// SpanIndexEncoder_34883724378827
// MI455X (gfx1250) — matrix operand order repaired and checked by reading, not compiled, not run on hardware
//
#include <hip/hip_runtime.h>

// ---------------------------------------------------------------------------
// SpanIndexEncoder as a masked GEMM on CDNA5 WMMA (gfx1250, wave32).
//   out[8192,256] = mask[8192,8192] @ emb[8192,256]   (fp32)
// mask generated in-register from span data; emb split into bf16 hi/lo so two
// v_wmma_f32_16x16x32_bf16 per tile reproduce fp32 accuracy.
// Fast path: prep kernel writes chunk-transposed bf16 hi/lo planes to d_ws;
// GEMM kernel streams them into double-buffered LDS with
// global_load_async_to_lds_b128 (ASYNCcnt), overlapping copy with WMMA.
// Each wave computes a 32x64 output tile so every B fragment feeds 4 WMMAs.
// ---------------------------------------------------------------------------

#define T_TOKENS 8192
#define N_NODES  8192
#define FEATS    256
#define KC       32               // nodes (K) per staged chunk
#define MT       32               // tokens per workgroup (fallback kernel)
#define MTA      64               // tokens per workgroup (async kernel)
#define THREADS  256              // 8 wave32
#define NCHUNK   (N_NODES / KC)   // 256
#define PLANE    (FEATS * KC)     // ushorts per plane (8192 = 16 KB)
#define CHUNK_US (2 * PLANE)      // ushorts per ws chunk block (hi + lo)

typedef __attribute__((ext_vector_type(16))) __bf16       v16bf;
typedef __attribute__((ext_vector_type(8)))  float        v8f;
typedef __attribute__((ext_vector_type(8)))  unsigned int v8u;

static __device__ __forceinline__ unsigned short f32_to_bf16_rne(float x) {
    unsigned u = __float_as_uint(x);
    u += 0x7FFFu + ((u >> 16) & 1u);          // round-to-nearest-even
    return (unsigned short)(u >> 16);
}

// ---------------------------------------------------------------------------
// Prep: emb f32 [8192][256] -> per-chunk bf16 hi/lo planes, transposed [f][k].
// ---------------------------------------------------------------------------
__global__ __launch_bounds__(THREADS)
void prep_bf16_hilo(const float* __restrict__ emb,
                    unsigned short* __restrict__ ws)
{
    __shared__ __align__(16) unsigned short sHi[PLANE];
    __shared__ __align__(16) unsigned short sLo[PLANE];
    const int tid = threadIdx.x;
    const int kc  = blockIdx.x * KC;

    #pragma unroll
    for (int i = 0; i < 8; ++i) {
        int q  = i * THREADS + tid;   // float4 index in [32][64]
        int k  = q >> 6;
        int f4 = q & 63;
        const float4 v = ((const float4*)(emb + (size_t)(kc + k) * FEATS))[f4];
        float xs[4] = {v.x, v.y, v.z, v.w};
        #pragma unroll
        for (int j = 0; j < 4; ++j) {
            int f = f4 * 4 + j;
            unsigned short hi = f32_to_bf16_rne(xs[j]);
            float hif = __uint_as_float(((unsigned)hi) << 16);
            unsigned short lo = f32_to_bf16_rne(xs[j] - hif);
            sHi[f * KC + k] = hi;
            sLo[f * KC + k] = lo;
        }
    }
    __syncthreads();

    uint4* dst = (uint4*)(ws + (size_t)blockIdx.x * CHUNK_US);
    const uint4* srcHi = (const uint4*)sHi;
    const uint4* srcLo = (const uint4*)sLo;
    #pragma unroll
    for (int i = 0; i < 4; ++i) {                     // 16 KB per plane
        dst[i * THREADS + tid]             = srcHi[i * THREADS + tid];
        dst[PLANE / 8 + i * THREADS + tid] = srcLo[i * THREADS + tid];
    }
}

// ---------------------------------------------------------------------------
// Fast GEMM: double-buffered async global->LDS staging + WMMA, 32x64 per wave.
// ---------------------------------------------------------------------------
__global__ __launch_bounds__(THREADS)
void span_encoder_wmma_async(const unsigned short* __restrict__ wsB,
                             const int*   __restrict__ starts,
                             const int*   __restrict__ ends,
                             const int*   __restrict__ numNodesPtr,
                             float*       __restrict__ out)
{
    __shared__ __align__(16) unsigned short sB[2][CHUNK_US];   // 64 KB
    __shared__ __align__(16) int sStart[KC];
    __shared__ __align__(16) int sEnd[KC];

    const int tid  = threadIdx.x;
    const int wave = tid >> 5;
    const int lane = tid & 31;
    const int hl   = lane >> 4;
    const int l16  = lane & 15;

    const int tokBase = blockIdx.x * MTA;
    const int mRow    = (wave & 1) * 32;        // wave's 32-row block
    const int fCol    = (wave >> 1) * 64;       // wave's 64-feature strip

    const int numNodes = numNodesPtr[0];
    const int tok0     = tokBase + mRow + l16;  // A fragment 0 row
    const int tok1     = tok0 + 16;             // A fragment 1 row

    // LDS byte offsets (generic LDS pointer: low 32 bits = LDS byte address)
    const unsigned ldsBase = (unsigned)(uintptr_t)&sB[0][0];

    // Issue one chunk's async copies: 32 KB = 2048 x b128, 8 per thread.
    auto issueChunk = [&](int c, int buf) {
        const unsigned long long g0 =
            (unsigned long long)(uintptr_t)(wsB + (size_t)c * CHUNK_US);
        const unsigned l0 = ldsBase + (unsigned)buf * (CHUNK_US * 2);
        #pragma unroll
        for (int i = 0; i < 8; ++i) {
            const unsigned off = (unsigned)(i * THREADS + tid) * 16u;
            const unsigned long long ga = g0 + off;
            const unsigned la = l0 + off;
            asm volatile("global_load_async_to_lds_b128 %0, %1, off"
                         :: "v"(la), "v"(ga) : "memory");
        }
    };

    v8f acc0[4] = {v8f{}, v8f{}, v8f{}, v8f{}};
    v8f acc1[4] = {v8f{}, v8f{}, v8f{}, v8f{}};

    issueChunk(0, 0);
    issueChunk(1, 1);

    for (int i = 0; i < NCHUNK; ++i) {
        const int buf = i & 1;
        const int kc  = i * KC;

        // chunk i's copies completed (in-order) while i+1 may be in flight
        if (i + 1 < NCHUNK) asm volatile("s_wait_asynccnt 0x8" ::: "memory");
        else                asm volatile("s_wait_asynccnt 0x0" ::: "memory");

        // stage spans for chunk i (invalid nodes -> empty span)
        if (tid < KC) {
            int n = kc + tid;
            sStart[tid] = (n < numNodes) ? starts[n] : 0x7FFFFFFF;
        } else if (tid < 2 * KC) {
            int k = tid - KC;
            int n = kc + k;
            sEnd[k] = (n < numNodes) ? ends[n] : -1;
        }
        __syncthreads();   // chunk-i data + spans visible to all waves

        // ---- bulk-load span data this lane needs: 8x ds_load_b128 ----
        // ss[i]   = sStart[8*hl + i]       (i = 0..7)
        // ss[8+i] = sStart[16 + 8*hl + i]
        int ss[16], ee[16];
        {
            const int b0 = 8 * hl;
            *(int4*)&ss[0]  = *(const int4*)&sStart[b0];
            *(int4*)&ss[4]  = *(const int4*)&sStart[b0 + 4];
            *(int4*)&ss[8]  = *(const int4*)&sStart[b0 + 16];
            *(int4*)&ss[12] = *(const int4*)&sStart[b0 + 20];
            *(int4*)&ee[0]  = *(const int4*)&sEnd[b0];
            *(int4*)&ee[4]  = *(const int4*)&sEnd[b0 + 4];
            *(int4*)&ee[8]  = *(const int4*)&sEnd[b0 + 16];
            *(int4*)&ee[12] = *(const int4*)&sEnd[b0 + 20];
        }

        // ---- A fragments: two 16x32 bf16 masks, branch-free VALU ----
        v8u a0raw, a1raw;
        #pragma unroll
        for (int p = 0; p < 8; ++p) {
            const int q0 = 2 * p, q1 = 2 * p + 1;
            unsigned m00 = ((ss[q0] <= tok0) & (tok0 <= ee[q0])) ? 0x3F80u : 0u;
            unsigned m01 = ((ss[q1] <= tok0) & (tok0 <= ee[q1])) ? 0x3F80u : 0u;
            unsigned m10 = ((ss[q0] <= tok1) & (tok1 <= ee[q0])) ? 0x3F80u : 0u;
            unsigned m11 = ((ss[q1] <= tok1) & (tok1 <= ee[q1])) ? 0x3F80u : 0u;
            a0raw[p] = m00 | (m01 << 16);
            a1raw[p] = m10 | (m11 << 16);
        }
        const v16bf a0 = __builtin_bit_cast(v16bf, a0raw);
        const v16bf a1 = __builtin_bit_cast(v16bf, a1raw);

        // ---- B fragments (hi & lo) -> 4 WMMAs each ----
        const unsigned short* bhiP = &sB[buf][0];
        const unsigned short* bloP = &sB[buf][PLANE];
        #pragma unroll
        for (int j = 0; j < 4; ++j) {
            const int f = fCol + j * 16 + l16;
            v8u bh, bl;
            #pragma unroll
            for (int p = 0; p < 8; ++p) {
                int k0 = 2 * p + 8 * hl + 8 * (p >> 2);   // even -> 4B-aligned packed pair
                bh[p] = *(const unsigned int*)&bhiP[f * KC + k0];
                bl[p] = *(const unsigned int*)&bloP[f * KC + k0];
            }
            const v16bf bhi = __builtin_bit_cast(v16bf, bh);
            const v16bf blo = __builtin_bit_cast(v16bf, bl);
            acc0[j] = __builtin_amdgcn_wmma_f32_16x16x32_bf16(
                          false, a0, false, bhi, (short)0, acc0[j],
                          false, false);
            acc0[j] = __builtin_amdgcn_wmma_f32_16x16x32_bf16(
                          false, a0, false, blo, (short)0, acc0[j],
                          false, false);
            acc1[j] = __builtin_amdgcn_wmma_f32_16x16x32_bf16(
                          false, a1, false, bhi, (short)0, acc1[j],
                          false, false);
            acc1[j] = __builtin_amdgcn_wmma_f32_16x16x32_bf16(
                          false, a1, false, blo, (short)0, acc1[j],
                          false, false);
        }
        __syncthreads();   // all reads of sB[buf] done

        if (i + 2 < NCHUNK) issueChunk(i + 2, buf);   // refill freed buffer
    }

    // ---- store C: VGPR r -> M = r + 8*half, N = lane%16 ----
    #pragma unroll
    for (int j = 0; j < 4; ++j) {
        const int f = fCol + j * 16 + l16;
        #pragma unroll
        for (int r = 0; r < 8; ++r) {
            const int m0 = mRow + r + 8 * hl;
            out[(size_t)(tokBase + m0) * FEATS + f]      = acc0[j][r];
            out[(size_t)(tokBase + m0 + 16) * FEATS + f] = acc1[j][r];
        }
    }
}

// ---------------------------------------------------------------------------
// Fallback fused kernel (no workspace needed).
// ---------------------------------------------------------------------------
__global__ __launch_bounds__(THREADS)
void span_encoder_wmma_fused(const float* __restrict__ emb,
                             const int*   __restrict__ starts,
                             const int*   __restrict__ ends,
                             const int*   __restrict__ numNodesPtr,
                             float*       __restrict__ out)
{
    __shared__ __align__(16) unsigned short sBhi[PLANE];
    __shared__ __align__(16) unsigned short sBlo[PLANE];
    __shared__ __align__(16) int sStart[KC];
    __shared__ __align__(16) int sEnd[KC];

    const int tid  = threadIdx.x;
    const int wave = tid >> 5;
    const int lane = tid & 31;
    const int hl   = lane >> 4;
    const int l16  = lane & 15;

    const int tokBase = blockIdx.x * MT;
    const int mRow    = (wave & 1) * 16;
    const int fCol    = (wave >> 1) * 64;

    const int numNodes = numNodesPtr[0];
    const int tok      = tokBase + mRow + l16;

    v8f acc[4] = {v8f{}, v8f{}, v8f{}, v8f{}};

    for (int kc = 0; kc < N_NODES; kc += KC) {
        if (tid < KC) {
            int n = kc + tid;
            sStart[tid] = (n < numNodes) ? starts[n] : 0x7FFFFFFF;
        } else if (tid < 2 * KC) {
            int k = tid - KC;
            int n = kc + k;
            sEnd[k] = (n < numNodes) ? ends[n] : -1;
        }
        #pragma unroll
        for (int i = 0; i < 8; ++i) {
            int q  = i * THREADS + tid;
            int k  = q >> 6;
            int f4 = q & 63;
            const float4 v =
                ((const float4*)(emb + (size_t)(kc + k) * FEATS))[f4];
            float xs[4] = {v.x, v.y, v.z, v.w};
            #pragma unroll
            for (int j = 0; j < 4; ++j) {
                int f = f4 * 4 + j;
                unsigned short hi = f32_to_bf16_rne(xs[j]);
                float hif = __uint_as_float(((unsigned)hi) << 16);
                unsigned short lo = f32_to_bf16_rne(xs[j] - hif);
                sBhi[f * KC + k] = hi;
                sBlo[f * KC + k] = lo;
            }
            if (kc + KC < N_NODES)
                __builtin_prefetch(emb + (size_t)(kc + KC + k) * FEATS + f4 * 4,
                                   0, 3);
        }
        __syncthreads();

        int ss[16], ee[16];
        {
            const int b0 = 8 * hl;
            *(int4*)&ss[0]  = *(const int4*)&sStart[b0];
            *(int4*)&ss[4]  = *(const int4*)&sStart[b0 + 4];
            *(int4*)&ss[8]  = *(const int4*)&sStart[b0 + 16];
            *(int4*)&ss[12] = *(const int4*)&sStart[b0 + 20];
            *(int4*)&ee[0]  = *(const int4*)&sEnd[b0];
            *(int4*)&ee[4]  = *(const int4*)&sEnd[b0 + 4];
            *(int4*)&ee[8]  = *(const int4*)&sEnd[b0 + 16];
            *(int4*)&ee[12] = *(const int4*)&sEnd[b0 + 20];
        }
        v8u araw;
        #pragma unroll
        for (int p = 0; p < 8; ++p) {
            const int q0 = 2 * p, q1 = 2 * p + 1;
            unsigned m0 = ((ss[q0] <= tok) & (tok <= ee[q0])) ? 0x3F80u : 0u;
            unsigned m1 = ((ss[q1] <= tok) & (tok <= ee[q1])) ? 0x3F80u : 0u;
            araw[p] = m0 | (m1 << 16);
        }
        const v16bf afrag = __builtin_bit_cast(v16bf, araw);

        #pragma unroll
        for (int j = 0; j < 4; ++j) {
            const int f = fCol + j * 16 + l16;
            v8u bh, bl;
            #pragma unroll
            for (int p = 0; p < 8; ++p) {
                int k0 = 2 * p + 8 * hl + 8 * (p >> 2);
                bh[p] = *(const unsigned int*)&sBhi[f * KC + k0];
                bl[p] = *(const unsigned int*)&sBlo[f * KC + k0];
            }
            const v16bf bhi = __builtin_bit_cast(v16bf, bh);
            const v16bf blo = __builtin_bit_cast(v16bf, bl);
            acc[j] = __builtin_amdgcn_wmma_f32_16x16x32_bf16(
                         false, afrag, false, bhi, (short)0, acc[j],
                         false, false);
            acc[j] = __builtin_amdgcn_wmma_f32_16x16x32_bf16(
                         false, afrag, false, blo, (short)0, acc[j],
                         false, false);
        }
        __syncthreads();
    }

    #pragma unroll
    for (int j = 0; j < 4; ++j) {
        const int f = fCol + j * 16 + l16;
        #pragma unroll
        for (int r = 0; r < 8; ++r) {
            const int m = mRow + r + 8 * hl;
            out[(size_t)(tokBase + m) * FEATS + f] = acc[j][r];
        }
    }
}

extern "C" void kernel_launch(void* const* d_in, const int* in_sizes, int n_in,
                              void* d_out, int out_size, void* d_ws, size_t ws_size,
                              hipStream_t stream) {
    (void)in_sizes; (void)n_in; (void)out_size;
    const float* emb      = (const float*)d_in[0];
    const int*   starts   = (const int*)d_in[1];
    const int*   ends     = (const int*)d_in[2];
    const int*   numNodes = (const int*)d_in[3];
    float*       out      = (float*)d_out;

    const size_t needed = (size_t)NCHUNK * CHUNK_US * sizeof(unsigned short); // 8 MB
    if (d_ws != nullptr && ws_size >= needed) {
        unsigned short* wsB = (unsigned short*)d_ws;
        prep_bf16_hilo<<<NCHUNK, THREADS, 0, stream>>>(emb, wsB);
        span_encoder_wmma_async<<<T_TOKENS / MTA, THREADS, 0, stream>>>(
            wsB, starts, ends, numNodes, out);
    } else {
        span_encoder_wmma_fused<<<T_TOKENS / MT, THREADS, 0, stream>>>(
            emb, starts, ends, numNodes, out);
    }
}
